// GQAttention_1941325218438
// MI455X (gfx1250) — matrix operand order repaired and checked by reading, not compiled, not run on hardware
//
#include <hip/hip_runtime.h>
#include <hip/hip_bf16.h>

typedef _Float16 half_t;
typedef __attribute__((ext_vector_type(16))) _Float16 v16h;
typedef __attribute__((ext_vector_type(8)))  _Float16 v8h;
typedef __attribute__((ext_vector_type(8)))  float    v8f;
typedef __attribute__((__vector_size__(4 * sizeof(int)))) int v4i;

#define WMMA_F16(a,b,c) __builtin_amdgcn_wmma_f32_16x16x32_f16(false,(a),false,(b),(short)0,(c),false,false)

static constexpr int BSZ = 4, T = 2048, DM = 1024, NH = 16, NKV = 4, HD = 64;
static constexpr int ROWS = BSZ * T;           // 8192
static constexpr int KVC  = NKV * HD;          // 256

// ---------------- async global->LDS (CDNA5), guarded with fallback ----------------
#if __has_builtin(__builtin_amdgcn_global_load_async_to_lds_b128)
#define HAS_ASYNC_LDS 1
#else
#define HAS_ASYNC_LDS 0
#endif

__device__ __forceinline__ void async_cp16(const half_t* g, half_t* l) {
#if HAS_ASYNC_LDS
  __builtin_amdgcn_global_load_async_to_lds_b128(
      (__attribute__((address_space(1))) v4i*)(g),
      (__attribute__((address_space(3))) v4i*)(l), 0, 0);
#else
  *(v8h*)l = *(const v8h*)g;
#endif
}

__device__ __forceinline__ void wait_async_upto8(bool more_inflight) {
#if HAS_ASYNC_LDS
#if __has_builtin(__builtin_amdgcn_s_wait_asynccnt)
  if (more_inflight) __builtin_amdgcn_s_wait_asynccnt(8);
  else               __builtin_amdgcn_s_wait_asynccnt(0);
#else
  if (more_inflight) asm volatile("s_wait_asynccnt 0x8" ::: "memory");
  else               asm volatile("s_wait_asynccnt 0x0" ::: "memory");
#endif
#endif
}

// ---------------- elementwise converts ----------------
__global__ void f32_to_f16_k(const float* __restrict__ s, half_t* __restrict__ d, int n) {
  int i = blockIdx.x * blockDim.x + threadIdx.x;
  int st = gridDim.x * blockDim.x;
  for (; i < n; i += st) d[i] = (half_t)s[i];
}

__global__ void f16_to_f32_k(const half_t* __restrict__ s, float* __restrict__ d, int n) {
  int i = blockIdx.x * blockDim.x + threadIdx.x;
  int st = gridDim.x * blockDim.x;
  for (; i < n; i += st) d[i] = (float)s[i];
}

// ---------------- RoPE (in-place on f16, optional f32 side output) ----------------
__global__ void rope_k(half_t* __restrict__ X, float* __restrict__ Fout,
                       const float* __restrict__ C, const float* __restrict__ S,
                       int rows, int ncols) {
  int ppr = ncols >> 1;                 // pairs per row
  long total = (long)rows * ppr;
  long i = blockIdx.x * (long)blockDim.x + threadIdx.x;
  long st = (long)gridDim.x * blockDim.x;
  for (; i < total; i += st) {
    int row = (int)(i / ppr), pc = (int)(i % ppr);
    int t = row & (T - 1);
    int p = pc & 31;                    // 32 pairs per head (HD=64)
    float c = C[t * 32 + p], s = S[t * 32 + p];
    size_t o = (size_t)row * ncols + 2 * pc;
    float x0 = (float)X[o], x1 = (float)X[o + 1];
    float y0 = x0 * c - x1 * s;
    float y1 = x0 * s + x1 * c;
    X[o] = (half_t)y0; X[o + 1] = (half_t)y1;
    if (Fout) { Fout[o] = y0; Fout[o + 1] = y1; }
  }
}

// ---------------- generic WMMA GEMM: C[M,N] = A[M,K] * W[N,K]^T ----------------
// One wave computes a 32x64 tile: 2 A-fragments share each B-fragment.
template <bool F32OUT>
__global__ __launch_bounds__(256) void gemm_wmma_k(
    const half_t* __restrict__ A, const half_t* __restrict__ W,
    half_t* __restrict__ Ch, float* __restrict__ Cf, int M, int N, int K) {
  const int lane = threadIdx.x & 31;
  const int wid  = (blockIdx.x * blockDim.x + threadIdx.x) >> 5;
  const int ntn  = N >> 6;                       // tiles of 64 cols
  const int total = (M >> 5) * ntn;              // tiles of 32 rows
  if (wid >= total) return;                      // wave-uniform
  const int mt = wid / ntn, nt = wid % ntn;
  const int m = lane & 15, hi = lane >> 4;

  v8f acc[2][4] = {};
  const half_t* arow0 = A + (size_t)(mt * 32 + m) * K;
  const half_t* arow1 = A + (size_t)(mt * 32 + 16 + m) * K;
  for (int k0 = 0; k0 < K; k0 += 32) {
    __builtin_prefetch(arow0 + k0 + 64, 0, 1);
    __builtin_prefetch(arow1 + k0 + 64, 0, 1);
    v16h a[2];
#pragma unroll
    for (int r = 0; r < 2; ++r) {
      const half_t* ar = (r == 0) ? arow0 : arow1;
      v8h lo = *(const v8h*)(ar + k0 + 16 * hi);
      v8h hj = *(const v8h*)(ar + k0 + 16 * hi + 8);
#pragma unroll
      for (int i = 0; i < 8; ++i) { a[r][i] = lo[i]; a[r][i + 8] = hj[i]; }
    }
#pragma unroll
    for (int j = 0; j < 4; ++j) {
      const half_t* wrow = W + (size_t)(nt * 64 + j * 16 + m) * K + k0 + 16 * hi;
      __builtin_prefetch(wrow + 64, 0, 1);
      v16h b = *(const v16h*)wrow;
      acc[0][j] = WMMA_F16(a[0], b, acc[0][j]);
      acc[1][j] = WMMA_F16(a[1], b, acc[1][j]);
    }
  }
#pragma unroll
  for (int r = 0; r < 2; ++r)
#pragma unroll
    for (int j = 0; j < 4; ++j)
#pragma unroll
      for (int v = 0; v < 8; ++v) {
        size_t row = (size_t)(mt * 32 + r * 16 + v + 8 * hi);
        size_t col = (size_t)(nt * 64 + j * 16 + m);
        if (F32OUT) Cf[row * N + col] = acc[r][j][v];
        else        Ch[row * N + col] = (half_t)acc[r][j][v];
      }
}

// ---------------- flash attention: 1 wave per (b, head, 16-query tile) ----------------
__global__ __launch_bounds__(32) void flash_attn_k(
    const half_t* __restrict__ Qh, const half_t* __restrict__ Kh,
    const half_t* __restrict__ Vh, half_t* __restrict__ Oh) {
  __shared__ half_t vtile[2][32][72];  // double-buffered; padded vs bank conflicts
  __shared__ half_t ptile[16][32];

  const int lane = threadIdx.x;
  const int wg   = blockIdx.x;          // B*NH*(T/16) = 8192
  const int tq   = wg & 127;
  const int head = (wg >> 7) & 15;
  const int b    = wg >> 11;
  const int kvh  = head >> 2;           // N_REP = 4
  const int qt0  = tq * 16;
  const int m = lane & 15, hi = lane >> 4;
  const float scale = 0.125f;           // 1/sqrt(64)

  // Q A-fragments (16 x 64 -> two k-tiles of 32)
  const half_t* qrow = Qh + (size_t)(b * T + qt0 + m) * DM + head * HD;
  v16h qa[2];
#pragma unroll
  for (int kt = 0; kt < 2; ++kt) {
    v8h lo = *(const v8h*)(qrow + kt * 32 + 16 * hi);
    v8h hj = *(const v8h*)(qrow + kt * 32 + 16 * hi + 8);
#pragma unroll
    for (int i = 0; i < 8; ++i) { qa[kt][i] = lo[i]; qa[kt][i + 8] = hj[i]; }
  }

  float mrow[8], lrow[8];
  v8f oacc[4] = {};
#pragma unroll
  for (int v = 0; v < 8; ++v) { mrow[v] = -1e30f; lrow[v] = 0.f; }

  const half_t* Kbase = Kh + (size_t)(b * T) * KVC + kvh * HD;
  const half_t* Vbase = Vh + (size_t)(b * T) * KVC + kvh * HD;

  const int nblocks = (qt0 + 16 + 31) >> 5;   // 32-key blocks

  // prime the V pipeline: stage block 0 into slot 0 (8 x b128 per lane)
  {
    const half_t* vr = Vbase + (size_t)(lane)*KVC;
#pragma unroll
    for (int c = 0; c < 8; ++c) async_cp16(vr + c * 8, &vtile[0][lane][c * 8]);
  }

  for (int blk = 0; blk < nblocks; ++blk) {
    const int s0 = blk * 32;
    const int slot = blk & 1;
    const bool have_next = (blk + 1) < nblocks;
    if (have_next) {   // overlap next block's V copy with this block's S/P math
      const half_t* vr = Vbase + (size_t)(s0 + 32 + lane) * KVC;
#pragma unroll
      for (int c = 0; c < 8; ++c)
        async_cp16(vr + c * 8, &vtile[slot ^ 1][lane][c * 8]);
    }

    // S = Q * K^T for two 16-key subtiles
    float stile[2][8];
    float newmax[8];
#pragma unroll
    for (int v = 0; v < 8; ++v) newmax[v] = mrow[v];
#pragma unroll
    for (int st = 0; st < 2; ++st) {
      v8f s = {};
#pragma unroll
      for (int kt = 0; kt < 2; ++kt) {
        const half_t* kr = Kbase + (size_t)(s0 + st * 16 + m) * KVC + kt * 32 + 16 * hi;
        v16h bfr = *(const v16h*)kr;
        s = WMMA_F16(qa[kt], bfr, s);
      }
      const int kk = s0 + st * 16 + m;           // key index held by this lane
#pragma unroll
      for (int v = 0; v < 8; ++v) {
        int qq = qt0 + v + 8 * hi;
        float val = s[v] * scale + ((kk > qq) ? -1e9f : 0.f);
        stile[st][v] = val;
        newmax[v] = fmaxf(newmax[v], val);
      }
    }
    // row-max across the 16 lanes of each C-fragment half (xor<16 stays in-group)
#pragma unroll
    for (int v = 0; v < 8; ++v) {
      float t0 = newmax[v];
      for (int msk = 1; msk < 16; msk <<= 1) t0 = fmaxf(t0, __shfl_xor(t0, msk, 32));
      newmax[v] = t0;
    }
    // rescale accumulators, compute P, stage P into LDS (C-layout -> row-major)
    float rsum[8];
#pragma unroll
    for (int v = 0; v < 8; ++v) {
      float fac = __expf(mrow[v] - newmax[v]);
      lrow[v] *= fac;
#pragma unroll
      for (int j = 0; j < 4; ++j) oacc[j][v] *= fac;
      mrow[v] = newmax[v];
      float p0 = __expf(stile[0][v] - newmax[v]);
      float p1 = __expf(stile[1][v] - newmax[v]);
      rsum[v] = p0 + p1;
      ptile[v + 8 * hi][m]      = (half_t)p0;
      ptile[v + 8 * hi][16 + m] = (half_t)p1;
    }
#pragma unroll
    for (int v = 0; v < 8; ++v) {
      float t0 = rsum[v];
      for (int msk = 1; msk < 16; msk <<= 1) t0 += __shfl_xor(t0, msk, 32);
      lrow[v] += t0;
    }

    wait_async_upto8(have_next);  // current V slot fully landed in LDS
    __syncthreads();              // dscnt wait: cross-lane P/V visibility

    // P as A-fragment (16x32)
    v16h pa;
    {
      v8h lo = *(const v8h*)(&ptile[m][16 * hi]);
      v8h hj = *(const v8h*)(&ptile[m][16 * hi + 8]);
#pragma unroll
      for (int i = 0; i < 8; ++i) { pa[i] = lo[i]; pa[i + 8] = hj[i]; }
    }
    // V B-fragments from LDS, O += P * V
#pragma unroll
    for (int j = 0; j < 4; ++j) {
      v16h vb;
#pragma unroll
      for (int i = 0; i < 16; ++i) vb[i] = vtile[slot][i + 16 * hi][j * 16 + m];
      oacc[j] = WMMA_F16(pa, vb, oacc[j]);
    }
    __syncthreads();              // order LDS reads before slot reuse
  }

  float invl[8];
#pragma unroll
  for (int v = 0; v < 8; ++v) invl[v] = 1.f / lrow[v];
#pragma unroll
  for (int j = 0; j < 4; ++j)
#pragma unroll
    for (int v = 0; v < 8; ++v) {
      size_t row = (size_t)(b * T + qt0 + v + 8 * hi);
      Oh[row * DM + head * HD + j * 16 + m] = (half_t)(oacc[j][v] * invl[v]);
    }
}

// ---------------- launcher ----------------
extern "C" void kernel_launch(void* const* d_in, const int* in_sizes, int n_in,
                              void* d_out, int out_size, void* d_ws, size_t ws_size,
                              hipStream_t stream) {
  (void)in_sizes; (void)n_in; (void)out_size; (void)ws_size;
  const float* x  = (const float*)d_in[0];
  const float* wq = (const float*)d_in[1];
  const float* wk = (const float*)d_in[2];
  const float* wv = (const float*)d_in[3];
  const float* wo = (const float*)d_in[4];
  const float* fc = (const float*)d_in[5];
  const float* fs = (const float*)d_in[6];
  // d_in[7] (mask) unused: causal mask computed analytically

  float* out = (float*)d_out;                       // [8192,1024]
  float* nk  = out + (size_t)ROWS * DM;             // new_k [8192,256]
  float* nv  = nk + (size_t)ROWS * KVC;             // new_v [8192,256]

  char* w = (char*)d_ws;
  auto take = [&](size_t halves) { half_t* p = (half_t*)w; w += halves * sizeof(half_t); return p; };
  half_t* xh  = take((size_t)ROWS * DM);
  half_t* wqh = take((size_t)DM * DM);
  half_t* wkh = take((size_t)KVC * DM);
  half_t* wvh = take((size_t)KVC * DM);
  half_t* woh = take((size_t)DM * DM);
  half_t* Qh  = take((size_t)ROWS * DM);
  half_t* Kh  = take((size_t)ROWS * KVC);
  half_t* Vh  = take((size_t)ROWS * KVC);
  half_t* Ah  = take((size_t)ROWS * DM);            // attention output (pre-projection)

  const int TPB = 256;
  // fp32 -> fp16 converts
  f32_to_f16_k<<<4096, TPB, 0, stream>>>(x,  xh,  ROWS * DM);
  f32_to_f16_k<<<2048, TPB, 0, stream>>>(wq, wqh, DM * DM);
  f32_to_f16_k<<<512,  TPB, 0, stream>>>(wk, wkh, KVC * DM);
  f32_to_f16_k<<<512,  TPB, 0, stream>>>(wv, wvh, KVC * DM);
  f32_to_f16_k<<<2048, TPB, 0, stream>>>(wo, woh, DM * DM);

  // Q/K/V projections (f16 out); 32x64 tiles -> (M/32)*(N/64) waves
  {
    int tiles = (ROWS / 32) * (DM / 64);            // 4096
    gemm_wmma_k<false><<<(tiles + 7) / 8, TPB, 0, stream>>>(xh, wqh, Qh, nullptr, ROWS, DM, DM);
    tiles = (ROWS / 32) * (KVC / 64);               // 1024
    gemm_wmma_k<false><<<(tiles + 7) / 8, TPB, 0, stream>>>(xh, wkh, Kh, nullptr, ROWS, KVC, DM);
    gemm_wmma_k<false><<<(tiles + 7) / 8, TPB, 0, stream>>>(xh, wvh, Vh, nullptr, ROWS, KVC, DM);
  }

  // RoPE (Q in-place; K in-place + fp32 new_k), new_v copy
  rope_k<<<4096, TPB, 0, stream>>>(Qh, nullptr, fc, fs, ROWS, DM);
  rope_k<<<2048, TPB, 0, stream>>>(Kh, nk,      fc, fs, ROWS, KVC);
  f16_to_f32_k<<<2048, TPB, 0, stream>>>(Vh, nv, ROWS * KVC);

  // flash attention: one wave per (b, head, 16-query tile)
  flash_attn_k<<<BSZ * NH * (T / 16), 32, 0, stream>>>(Qh, Kh, Vh, Ah);

  // output projection (f32 out straight to d_out)
  {
    int tiles = (ROWS / 32) * (DM / 64);            // 4096
    gemm_wmma_k<true><<<(tiles + 7) / 8, TPB, 0, stream>>>(Ah, woh, nullptr, out, ROWS, DM, DM);
  }
}
